// SatGNN_70531952935340
// MI455X (gfx1250) — matrix operand order repaired and checked by reading, not compiled, not run on hardware
//
#include <hip/hip_runtime.h>
#include <hip/hip_bf16.h>

#define HDIM 64

typedef __attribute__((ext_vector_type(16))) __bf16 v16bf;
typedef __attribute__((ext_vector_type(8)))  float  v8f;

// ---------------------------------------------------------------------------
// Weight pre-swizzle: pack a 64x64 f32 row-major W into bf16 B-fragments in
// the exact 16-bit B-matrix VGPR layout (ISA 7.12.2): for fragment (nt, kh),
// lane l holds 16 contiguous bf16 = W[kbase+(j&7)+(j>>3)*16][n], n=(l&15)+nt*16,
// kbase = kh*32 + (l>=16)*8.  Output index: ((nt*2+kh)*32 + lane)*16 + j.
// Each wave's fragment load becomes one 32B contiguous vector load.
// ---------------------------------------------------------------------------
__global__ __launch_bounds__(256) void wswz_kernel(const float* __restrict__ W,
                                                   __bf16* __restrict__ Wsw)
{
    int t = blockIdx.x * blockDim.x + threadIdx.x;   // 0..4095
    if (t >= 4096) return;
    int j    = t & 15;
    int lane = (t >> 4) & 31;
    int kh   = (t >> 9) & 1;
    int nt   = (t >> 10) & 3;
    int n     = (lane & 15) + nt * 16;
    int kbase = kh * 32 + (lane >> 4) * 8;
    Wsw[t] = (__bf16)W[(size_t)(kbase + (j & 7) + (j >> 3) * 16) * HDIM + n];
}

// ---------------------------------------------------------------------------
// A-fragment: 16x32 bf16 (M x K), 16-bit A layout. Per lane this is two
// contiguous 8-float runs of the row: K = base..base+7 and base+16..base+23,
// base = kh*32 + (lane>=16)*8  ->  4x b128 loads + packed cvt.
// Caller guarantees the whole 16-row tile is in bounds (nrows % 16 == 0).
// ---------------------------------------------------------------------------
__device__ __forceinline__ v16bf load_a_frag(const float* __restrict__ X, int row0,
                                             int kh, float scale, int lane) {
    int r  = row0 + (lane & 15);
    int hi = lane >> 4;
    const float* xr = X + (size_t)r * HDIM + kh * 32 + hi * 8;
    float4 p0 = *(const float4*)(xr);
    float4 p1 = *(const float4*)(xr + 4);
    float4 p2 = *(const float4*)(xr + 16);
    float4 p3 = *(const float4*)(xr + 20);
    v16bf a;
    a[0]  = (__bf16)(p0.x * scale); a[1]  = (__bf16)(p0.y * scale);
    a[2]  = (__bf16)(p0.z * scale); a[3]  = (__bf16)(p0.w * scale);
    a[4]  = (__bf16)(p1.x * scale); a[5]  = (__bf16)(p1.y * scale);
    a[6]  = (__bf16)(p1.z * scale); a[7]  = (__bf16)(p1.w * scale);
    a[8]  = (__bf16)(p2.x * scale); a[9]  = (__bf16)(p2.y * scale);
    a[10] = (__bf16)(p2.z * scale); a[11] = (__bf16)(p2.w * scale);
    a[12] = (__bf16)(p3.x * scale); a[13] = (__bf16)(p3.y * scale);
    a[14] = (__bf16)(p3.z * scale); a[15] = (__bf16)(p3.w * scale);
    return a;
}

__device__ __forceinline__ v16bf load_b_frag(const __bf16* __restrict__ Wsw,
                                             int nt, int kh, int lane) {
    return *(const v16bf*)(Wsw + (size_t)(((nt * 2 + kh) * 32 + lane) * 16));
}

// ---------------------------------------------------------------------------
// Y[n,64] = relu( s1*(X1@W1) + (X2@W2 if DUAL) + bscale*bias )
// One wave = 16 rows x 64 cols: 4 N-tiles x 2 K-halves -> 8 (16 dual) WMMAs.
// Wave-level row guard only; EXEC stays all-ones around every v_wmma and the
// epilogue stores are unconditional (nrows % 16 == 0 for every call site).
// ---------------------------------------------------------------------------
template <bool DUAL>
__global__ __launch_bounds__(256) void gemm64_kernel(
    const float* __restrict__ X1, const __bf16* __restrict__ W1, float s1,
    const float* __restrict__ X2, const __bf16* __restrict__ W2,
    const float* __restrict__ bias, float bscale,
    float* __restrict__ Y, int nrows)
{
    int lane = threadIdx.x & 31;
    int wave = threadIdx.x >> 5;
    int row0 = (blockIdx.x * 8 + wave) * 16;
    if (row0 >= nrows) return;   // uniform across wave

    v16bf a1_0 = load_a_frag(X1, row0, 0, s1, lane);
    v16bf a1_1 = load_a_frag(X1, row0, 1, s1, lane);
    v16bf a2_0, a2_1;
    if (DUAL) {
        a2_0 = load_a_frag(X2, row0, 0, 1.0f, lane);
        a2_1 = load_a_frag(X2, row0, 1, 1.0f, lane);
    }

    int hi = lane >> 4;
    float* yb = Y + (size_t)(row0 + hi * 8) * HDIM + (lane & 15);
#pragma unroll
    for (int nt = 0; nt < 4; ++nt) {
        float bv = bias[(lane & 15) + nt * 16] * bscale;
        v8f acc;
#pragma unroll
        for (int v = 0; v < 8; ++v) acc[v] = bv;

        v16bf b0 = load_b_frag(W1, nt, 0, lane);
        v16bf b1 = load_b_frag(W1, nt, 1, lane);
        acc = __builtin_amdgcn_wmma_f32_16x16x32_bf16(false, a1_0, false, b0,
                                                      (short)0, acc, false, false);
        acc = __builtin_amdgcn_wmma_f32_16x16x32_bf16(false, a1_1, false, b1,
                                                      (short)0, acc, false, false);
        if (DUAL) {
            v16bf c0 = load_b_frag(W2, nt, 0, lane);
            v16bf c1 = load_b_frag(W2, nt, 1, lane);
            acc = __builtin_amdgcn_wmma_f32_16x16x32_bf16(false, a2_0, false, c0,
                                                          (short)0, acc, false, false);
            acc = __builtin_amdgcn_wmma_f32_16x16x32_bf16(false, a2_1, false, c1,
                                                          (short)0, acc, false, false);
        }
        // D layout: VGPR v -> row row0+v (lanes 0-15) / row0+v+8 (lanes 16-31)
#pragma unroll
        for (int v = 0; v < 8; ++v)
            yb[(size_t)v * HDIM + nt * 16] = fmaxf(acc[v], 0.0f);
    }
}

// ---------------------------------------------------------------------------
// Small-K input embedding: h = relu(x @ W + b), one thread per (row, col).
// ---------------------------------------------------------------------------
template <int K>
__global__ __launch_bounds__(256) void embed_kernel(const float* __restrict__ X,
                                                    const float* __restrict__ W,
                                                    const float* __restrict__ b,
                                                    float* __restrict__ Y, int n)
{
    int idx = blockIdx.x * blockDim.x + threadIdx.x;
    int row = idx >> 6, col = idx & 63;
    if (row >= n) return;
    float s = b[col];
#pragma unroll
    for (int k = 0; k < K; ++k) s += X[(size_t)row * K + k] * W[k * HDIM + col];
    Y[(size_t)row * HDIM + col] = fmaxf(s, 0.0f);
}

// ---------------------------------------------------------------------------
// segment_max scatter. pooled >= 0 (post-relu) and agg init = 0, so integer
// atomic max on the IEEE bit pattern == float max, and zero in-degree -> 0,
// matching jnp.where(isfinite(segment_max), ..., 0). One wave per edge;
// each lane moves a float2 (256B/row, fully coalesced).
// ---------------------------------------------------------------------------
__global__ __launch_bounds__(256) void edge_max_kernel(const float* __restrict__ pooled,
                                                       const int* __restrict__ src,
                                                       const int* __restrict__ dst,
                                                       float* __restrict__ agg, int nedges)
{
    int t = blockIdx.x * blockDim.x + threadIdx.x;
    int e = t >> 5;
    if (e >= nedges) return;     // uniform across wave
    int lane = t & 31;
    int s = src[e], d = dst[e];
    float2 v = *(const float2*)(pooled + (size_t)s * HDIM + lane * 2);
    int* out = (int*)(agg + (size_t)d * HDIM + lane * 2);
    atomicMax(out,     __float_as_int(v.x));
    atomicMax(out + 1, __float_as_int(v.y));
}

__global__ __launch_bounds__(256) void add_inplace_kernel(float* __restrict__ a,
                                                          const float* __restrict__ b,
                                                          size_t n)
{
    size_t i = (size_t)blockIdx.x * blockDim.x + threadIdx.x;
    if (i < n) a[i] += b[i];
}

// out = z @ W_out(64x1) + b_out
__global__ __launch_bounds__(256) void out_kernel(const float* __restrict__ z,
                                                  const float* __restrict__ Wout,
                                                  const float* __restrict__ bout,
                                                  float* __restrict__ out, int n)
{
    int r = blockIdx.x * blockDim.x + threadIdx.x;
    if (r >= n) return;
    float s = bout[0];
#pragma unroll
    for (int k = 0; k < HDIM; ++k) s += z[(size_t)r * HDIM + k] * Wout[k];
    out[r] = s;
}

// ---------------------------------------------------------------------------

extern "C" void kernel_launch(void* const* d_in, const int* in_sizes, int n_in,
                              void* d_out, int out_size, void* d_ws, size_t ws_size,
                              hipStream_t stream)
{
    const int NV = in_sizes[0] / 7;   // 100000  (multiple of 16)
    const int NC = in_sizes[1] / 4;   // 100000  (multiple of 16)
    const int NS = in_sizes[2] / 6;   // 20000   (multiple of 16)
    const int EVC = in_sizes[33];     // 2,000,000
    const int ESC = in_sizes[35];     // 300,000

    const float* var_x = (const float*)d_in[0];
    const float* con_x = (const float*)d_in[1];
    const float* soc_x = (const float*)d_in[2];
    const float* W_var = (const float*)d_in[3]; const float* b_var = (const float*)d_in[4];
    const float* W_con = (const float*)d_in[5]; const float* b_con = (const float*)d_in[6];
    const float* W_soc = (const float*)d_in[7]; const float* b_soc = (const float*)d_in[8];
    const int L1F = 9, L1B = 14, L2F = 19, L2B = 24;   // {Wp, bp, Ws, Wn, b}
    const float* W_hid = (const float*)d_in[29]; const float* b_hid = (const float*)d_in[30];
    const float* W_out = (const float*)d_in[31]; const float* b_out = (const float*)d_in[32];
    const int* vc_src = (const int*)d_in[33];
    const int* vc_dst = (const int*)d_in[34];
    const int* sc_src = (const int*)d_in[35];
    const int* sc_dst = (const int*)d_in[36];

    // -------- workspace carving --------
    // 13 swizzled bf16 weight slots (4096 bf16 = 8KB each) first, then f32 arrays.
    const size_t WSLOT = 4096;
    __bf16* wsw = (__bf16*)d_ws;
    // slots: 0-2 l1f{Wp,Ws,Wn}, 3-5 l1b, 6-8 l2f, 9-11 l2b, 12 W_hid
    float* p = (float*)((char*)d_ws + 13 * WSLOT * sizeof(__bf16));   // 104KB, 16B aligned
    float* hv    = p; p += (size_t)NV * HDIM;
    float* hv2   = p; p += (size_t)NV * HDIM;
    float* hc    = p; p += (size_t)NC * HDIM;
    float* hc2   = p; p += (size_t)NC * HDIM;
    float* hs    = p; p += (size_t)NS * HDIM;
    float* hs2   = p; p += (size_t)NS * HDIM;
    float* poolA = p; p += (size_t)NV * HDIM;   // pooled_var / pooled_con (NV==NC)
    float* poolB = p; p += (size_t)NS * HDIM;   // pooled_soc
    float* aggA  = p; p += (size_t)NC * HDIM;   // v2c agg / c2v agg
    float* aggB  = p; p += (size_t)NC * HDIM;   // s2c agg / c2s agg

    // -------- pre-swizzle all 64x64 weights to bf16 fragment layout --------
    const int layer_bases[4] = {L1F, L1B, L2F, L2B};
    for (int li = 0; li < 4; ++li) {
        wswz_kernel<<<16, 256, 0, stream>>>((const float*)d_in[layer_bases[li] + 0],
                                            wsw + (size_t)(li * 3 + 0) * WSLOT);   // Wp
        wswz_kernel<<<16, 256, 0, stream>>>((const float*)d_in[layer_bases[li] + 2],
                                            wsw + (size_t)(li * 3 + 1) * WSLOT);   // Ws
        wswz_kernel<<<16, 256, 0, stream>>>((const float*)d_in[layer_bases[li] + 3],
                                            wsw + (size_t)(li * 3 + 2) * WSLOT);   // Wn
    }
    wswz_kernel<<<16, 256, 0, stream>>>(W_hid, wsw + (size_t)12 * WSLOT);

    auto gemm1 = [&](const float* X1, const __bf16* W1,
                     const float* bias, float* Y, int n) {
        gemm64_kernel<false><<<(n + 127) / 128, 256, 0, stream>>>(
            X1, W1, 1.0f, nullptr, nullptr, bias, 1.0f, Y, n);
    };
    auto gemm2 = [&](const float* X1, const __bf16* W1, float s1,
                     const float* X2, const __bf16* W2,
                     const float* bias, float bs, float* Y, int n) {
        gemm64_kernel<true><<<(n + 127) / 128, 256, 0, stream>>>(
            X1, W1, s1, X2, W2, bias, bs, Y, n);
    };

    // -------- input embeddings --------
    embed_kernel<7><<<((size_t)NV * 64 + 255) / 256, 256, 0, stream>>>(var_x, W_var, b_var, hv, NV);
    embed_kernel<4><<<((size_t)NC * 64 + 255) / 256, 256, 0, stream>>>(con_x, W_con, b_con, hc, NC);
    embed_kernel<6><<<((size_t)NS * 64 + 255) / 256, 256, 0, stream>>>(soc_x, W_soc, b_soc, hs, NS);

    // -------- forward half-passes: update constraints (l1f then l2f) --------
    const int fwd_slot[2] = {0, 6};     // wsw slot base for l1f, l2f
    const int fwd_base[2] = {L1F, L2F};
    for (int li = 0; li < 2; ++li) {
        const __bf16* WpS = wsw + (size_t)(fwd_slot[li] + 0) * WSLOT;
        const __bf16* WsS = wsw + (size_t)(fwd_slot[li] + 1) * WSLOT;
        const __bf16* WnS = wsw + (size_t)(fwd_slot[li] + 2) * WSLOT;
        const float* bp = (const float*)d_in[fwd_base[li] + 1];
        const float* bb = (const float*)d_in[fwd_base[li] + 4];

        gemm1(hv, WpS, bp, poolA, NV);  // relu(hv@Wp+bp)
        gemm1(hs, WpS, bp, poolB, NS);  // relu(hs@Wp+bp)

        hipMemsetAsync(aggA, 0, (size_t)NC * HDIM * sizeof(float), stream);
        hipMemsetAsync(aggB, 0, (size_t)NC * HDIM * sizeof(float), stream);
        edge_max_kernel<<<((size_t)EVC * 32 + 255) / 256, 256, 0, stream>>>(poolA, vc_src, vc_dst, aggA, EVC);
        edge_max_kernel<<<((size_t)ESC * 32 + 255) / 256, 256, 0, stream>>>(poolB, sc_src, sc_dst, aggB, ESC);
        add_inplace_kernel<<<((size_t)NC * HDIM + 255) / 256, 256, 0, stream>>>(aggA, aggB, (size_t)NC * HDIM);

        // shared-weight sum of two SAGEConvs: relu(2*(hc@Ws) + (aggv+aggs)@Wn + 2b)
        gemm2(hc, WsS, 2.0f, aggA, WnS, bb, 2.0f, hc2, NC);
        float* t = hc; hc = hc2; hc2 = t;
    }

    // -------- backward half-passes: update var & soc (l1b then l2b) --------
    const int bwd_slot[2] = {3, 9};
    const int bwd_base[2] = {L1B, L2B};
    for (int li = 0; li < 2; ++li) {
        const __bf16* WpS = wsw + (size_t)(bwd_slot[li] + 0) * WSLOT;
        const __bf16* WsS = wsw + (size_t)(bwd_slot[li] + 1) * WSLOT;
        const __bf16* WnS = wsw + (size_t)(bwd_slot[li] + 2) * WSLOT;
        const float* bp = (const float*)d_in[bwd_base[li] + 1];
        const float* bb = (const float*)d_in[bwd_base[li] + 4];

        gemm1(hc, WpS, bp, poolA, NC);  // pooled_con

        hipMemsetAsync(aggA, 0, (size_t)NV * HDIM * sizeof(float), stream);
        hipMemsetAsync(aggB, 0, (size_t)NS * HDIM * sizeof(float), stream);
        // reversed edges: src index = con side, segment = var/soc side
        edge_max_kernel<<<((size_t)EVC * 32 + 255) / 256, 256, 0, stream>>>(poolA, vc_dst, vc_src, aggA, EVC);
        edge_max_kernel<<<((size_t)ESC * 32 + 255) / 256, 256, 0, stream>>>(poolA, sc_dst, sc_src, aggB, ESC);

        gemm2(hv, WsS, 1.0f, aggA, WnS, bb, 1.0f, hv2, NV);
        gemm2(hs, WsS, 1.0f, aggB, WnS, bb, 1.0f, hs2, NS);
        float* t;
        t = hv; hv = hv2; hv2 = t;
        t = hs; hs = hs2; hs2 = t;
    }

    // -------- output MLP: same W_hid applied twice, then 64->1 head --------
    const __bf16* WhS = wsw + (size_t)12 * WSLOT;
    gemm1(hv, WhS, b_hid, poolA, NV);
    gemm1(poolA, WhS, b_hid, aggA, NV);
    out_kernel<<<(NV + 255) / 256, 256, 0, stream>>>(aggA, W_out, b_out, (float*)d_out, NV);
}
